// ODELSTM_58248346468500
// MI455X (gfx1250) — matrix operand order repaired and checked by reading, not compiled, not run on hardware
//
#include <hip/hip_runtime.h>
#include <hip/hip_bf16.h>

#define B_SZ 256
#define T_SZ 64
#define IN_SZ 256
#define H_SZ 512
#define OH_SZ 64
#define WG_ROWS 16
#define NTHREADS 256

typedef __attribute__((ext_vector_type(16))) __bf16 v16bf;
typedef __attribute__((ext_vector_type(8)))  __bf16 v8bf;
typedef __attribute__((ext_vector_type(8)))  float  v8f;
typedef __attribute__((ext_vector_type(4)))  float  v4f;

// Dormand-Prince 5(4): A rows (triangular, row r at offset r*(r+1)/2), b weights
__constant__ float c_DPA[15] = {
    1.f/5.f,
    3.f/40.f, 9.f/40.f,
    44.f/45.f, -56.f/15.f, 32.f/9.f,
    19372.f/6561.f, -25360.f/2187.f, 64448.f/6561.f, -212.f/729.f,
    9017.f/3168.f, -355.f/33.f, 46732.f/5247.f, 49.f/176.f, -5103.f/18656.f};
__constant__ float c_DPB[6] = {
    35.f/384.f, 0.f, 500.f/1113.f, 125.f/192.f, -2187.f/6784.f, 11.f/84.f};

// ---- bf16 <-> f32 via integer ops (RNE), no reliance on __bf16 arithmetic ----
__device__ __forceinline__ __bf16 f2bf(float f) {
    unsigned u = __builtin_bit_cast(unsigned, f);
    unsigned short s = (unsigned short)((u + 0x7FFFu + ((u >> 16) & 1u)) >> 16);
    return __builtin_bit_cast(__bf16, s);
}
__device__ __forceinline__ float bf2f(__bf16 b) {
    unsigned short s = __builtin_bit_cast(unsigned short, b);
    unsigned u = ((unsigned)s) << 16;
    return __builtin_bit_cast(float, u);
}
__device__ __forceinline__ float sigmoidf(float x) { return 1.f / (1.f + __expf(-x)); }

__device__ __forceinline__ v8f vzero8() {
    v8f z = {0.f, 0.f, 0.f, 0.f, 0.f, 0.f, 0.f, 0.f};
    return z;
}
__device__ __forceinline__ v8f wmma_bf16(v16bf a, v16bf b, v8f c) {
    return __builtin_amdgcn_wmma_f32_16x16x32_bf16(false, a, false, b, (short)0, c,
                                                   false, false);
}

// A fragment (16x32, M x K) from fp32 source, row stride ldk.
// lane<16: M=lane, K = {k0..k0+7, k0+16..k0+23}; lane>=16: M=lane-16, K shifted +8.
__device__ __forceinline__ v16bf load_A_f32(const float* base, int ldk, int k0, int lane) {
    const float* p = base + (lane & 15) * ldk + k0 + 8 * (lane >> 4);
    v4f f0 = *(const v4f*)p;
    v4f f1 = *(const v4f*)(p + 4);
    v4f f2 = *(const v4f*)(p + 16);
    v4f f3 = *(const v4f*)(p + 20);
    v16bf a;
#pragma unroll
    for (int e = 0; e < 4; ++e) {
        a[e]      = f2bf(f0[e]);
        a[e + 4]  = f2bf(f1[e]);
        a[e + 8]  = f2bf(f2[e]);
        a[e + 12] = f2bf(f3[e]);
    }
    return a;
}

// A fragment from bf16 source (LDS or global), same layout: two 16B vector loads.
__device__ __forceinline__ v16bf load_A_bf16(const __bf16* base, int ldk, int k0, int lane) {
    const __bf16* p = base + (lane & 15) * ldk + k0 + 8 * (lane >> 4);
    v8bf lo = *(const v8bf*)p;
    v8bf hi = *(const v8bf*)(p + 16);
    v16bf a;
#pragma unroll
    for (int e = 0; e < 8; ++e) {
        a[e]     = lo[e];
        a[e + 8] = hi[e];
    }
    return a;
}

// B fragment (32x16, K x N) for D = A @ W^T: B[k,n] = W[wr0+n, k0+k], W row-major
// with row length kdim. lane holds 16 bf16 of one weight row:
// lane<16: N=lane, K = {k0..k0+7, k0+16..k0+23}; lane>=16: N=lane-16, K shifted +8.
__device__ __forceinline__ v16bf load_B_w(const __bf16* W, int kdim, int wr0, int k0,
                                          int lane) {
    const __bf16* p = W + (size_t)(wr0 + (lane & 15)) * kdim + k0 + 8 * (lane >> 4);
    v8bf lo = *(const v8bf*)p;
    v8bf hi = *(const v8bf*)(p + 16);
    v16bf b;
#pragma unroll
    for (int e = 0; e < 8; ++e) {
        b[e]     = lo[e];
        b[e + 8] = hi[e];
    }
    return b;
}

// ---------------- weight pre-conversion ----------------
__global__ void cvt_bf16_kernel(const float* __restrict__ src, __bf16* __restrict__ dst,
                                int n) {
    int i = blockIdx.x * blockDim.x + threadIdx.x;
    if (i < n) dst[i] = f2bf(src[i]);
}
__global__ void bias_sum_kernel(const float* __restrict__ bi, const float* __restrict__ bh,
                                float* __restrict__ out, int n) {
    int i = blockIdx.x * blockDim.x + threadIdx.x;
    if (i < n) out[i] = bi[i] + bh[i];
}

// ---------------- fused ODE-LSTM: whole sequence, one kernel ----------------
__global__ __launch_bounds__(NTHREADS, 1) void odelstm_fused(
    const float* __restrict__ x, const float* __restrict__ tau,
    const __bf16* __restrict__ Wi0, const __bf16* __restrict__ Wh0,
    const __bf16* __restrict__ Wi1, const __bf16* __restrict__ Wh1,
    const __bf16* __restrict__ W1_0, const __bf16* __restrict__ W2_0,
    const __bf16* __restrict__ W1_1, const __bf16* __restrict__ W2_1,
    const float* __restrict__ bsum0, const float* __restrict__ bsum1,
    const float* __restrict__ b1_0, const float* __restrict__ b2_0,
    const float* __restrict__ b1_1, const float* __restrict__ b2_1,
    float* __restrict__ out, float* __restrict__ outHT, float* __restrict__ outCT) {
    extern __shared__ char smem[];
    const int NE = WG_ROWS * H_SZ;  // 8192 elements per state buffer
    float* c0 = (float*)smem;                       // fp32 cell state, layer 0
    float* c1 = c0 + NE;                            // fp32 cell state, layer 1
    float* s1 = c1 + NE;                            // fp32 h_new / ODE base & result
    __bf16* hbf0 = (__bf16*)(s1 + NE);              // bf16 h mirror, layer 0
    __bf16* hbf1 = hbf0 + NE;                       // bf16 h mirror, layer 1
    __bf16* s2 = hbf1 + NE;                         // bf16 dopri5 stage y
    __bf16* kbuf = s2 + NE;                         // bf16 6 stage slopes k1..k6
    __bf16* zbuf = kbuf + 6 * NE;                   // bf16 16 x 64 hidden of f()
    float* dts = (float*)(zbuf + WG_ROWS * OH_SZ);  // per-row dt

    const int tid = threadIdx.x;
    const int lane = tid & 31;
    const int wid = tid >> 5;
    const int half = lane >> 4;
    const int nlo = lane & 15;
    const int rbase = blockIdx.x * WG_ROWS;

    // zero-init recurrent state
    for (int i = tid; i < NE; i += NTHREADS) {
        c0[i] = 0.f;
        c1[i] = 0.f;
        hbf0[i] = __builtin_bit_cast(__bf16, (unsigned short)0);
        hbf1[i] = __builtin_bit_cast(__bf16, (unsigned short)0);
    }
    __syncthreads();

    for (int t = 0; t < T_SZ; ++t) {
        if (tid < WG_ROWS)
            dts[tid] = 0.5f * tau[(size_t)(rbase + tid) * T_SZ + t];  // tau / ODE_STEPS
        __syncthreads();

        for (int l = 0; l < 2; ++l) {
            const __bf16* Wi = l ? Wi1 : Wi0;
            const __bf16* Wh = l ? Wh1 : Wh0;
            const __bf16* W1 = l ? W1_1 : W1_0;
            const __bf16* W2 = l ? W2_1 : W2_0;
            const float* bsum = l ? bsum1 : bsum0;
            const float* b1 = l ? b1_1 : b1_0;
            const float* b2 = l ? b2_1 : b2_0;
            __bf16* hbfl = l ? hbf1 : hbf0;
            float* cl = l ? c1 : c0;

            // ---- LSTM cell: gates = x@Wi^T + h@Wh^T + (bi+bh), fused i/f/g/o ----
            // 16 live accumulator tiles: [col-block][gate]; one A load feeds 16 WMMAs.
            v8f acc[4][4];
#pragma unroll
            for (int nb = 0; nb < 4; ++nb)
#pragma unroll
                for (int g = 0; g < 4; ++g) acc[nb][g] = vzero8();

            if (l == 0) {
                const float* xg = x + (size_t)rbase * T_SZ * IN_SZ + (size_t)t * IN_SZ;
                for (int k0 = 0; k0 < IN_SZ; k0 += 32) {
                    v16bf a = load_A_f32(xg, T_SZ * IN_SZ, k0, lane);
#pragma unroll
                    for (int nb = 0; nb < 4; ++nb) {
                        const int n0 = (wid * 4 + nb) * 16;
#pragma unroll
                        for (int g = 0; g < 4; ++g)
                            acc[nb][g] = wmma_bf16(
                                a, load_B_w(Wi, IN_SZ, g * H_SZ + n0, k0, lane),
                                acc[nb][g]);
                    }
                }
            } else {
                for (int k0 = 0; k0 < H_SZ; k0 += 32) {
                    v16bf a = load_A_bf16(hbf0, H_SZ, k0, lane);
#pragma unroll
                    for (int nb = 0; nb < 4; ++nb) {
                        const int n0 = (wid * 4 + nb) * 16;
#pragma unroll
                        for (int g = 0; g < 4; ++g)
                            acc[nb][g] = wmma_bf16(
                                a, load_B_w(Wi, H_SZ, g * H_SZ + n0, k0, lane),
                                acc[nb][g]);
                    }
                }
            }
            for (int k0 = 0; k0 < H_SZ; k0 += 32) {
                if (k0 + 32 < H_SZ)
                    __builtin_prefetch(Wh + (size_t)(wid * 64) * H_SZ + k0 + 32, 0, 0);
                v16bf a = load_A_bf16(hbfl, H_SZ, k0, lane);
#pragma unroll
                for (int nb = 0; nb < 4; ++nb) {
                    const int n0 = (wid * 4 + nb) * 16;
#pragma unroll
                    for (int g = 0; g < 4; ++g)
                        acc[nb][g] = wmma_bf16(
                            a, load_B_w(Wh, H_SZ, g * H_SZ + n0, k0, lane), acc[nb][g]);
                }
            }
#pragma unroll
            for (int nb = 0; nb < 4; ++nb) {
                const int col = (wid * 4 + nb) * 16 + nlo;
#pragma unroll
                for (int r = 0; r < 8; ++r) {
                    const int m = r + 8 * half;
                    float gi = sigmoidf(acc[nb][0][r] + bsum[col]);
                    float gf = sigmoidf(acc[nb][1][r] + bsum[H_SZ + col]);
                    float gg = tanhf(acc[nb][2][r] + bsum[2 * H_SZ + col]);
                    float go = sigmoidf(acc[nb][3][r] + bsum[3 * H_SZ + col]);
                    float cn = gf * cl[m * H_SZ + col] + gi * gg;
                    cl[m * H_SZ + col] = cn;
                    s1[m * H_SZ + col] = go * tanhf(cn);  // ODE initial condition
                }
            }
            __syncthreads();

            // ---- ODE: 2 fixed dopri5 substeps, f(h)=tanh(h@W1^T+b1)@W2^T+b2 ----
            for (int sub = 0; sub < 2; ++sub) {
                for (int st = 0; st < 6; ++st) {
                    // y_stage = base + dt * sum_j A[st-1][j] * k_j   (bf16 into s2)
                    for (int i = tid; i < NE; i += NTHREADS) {
                        float y = s1[i];
                        if (st > 0) {
                            float dt = dts[i >> 9];
                            const int ao2 = (st - 1) * st / 2;
                            float a2 = 0.f;
                            for (int j = 0; j < st; ++j)
                                a2 += c_DPA[ao2 + j] * bf2f(kbuf[j * NE + i]);
                            y += dt * a2;
                        }
                        s2[i] = f2bf(y);
                    }
                    __syncthreads();
                    // z = tanh(y @ W1^T + b1): 16x64, waves 0..3
                    if (wid < 4) {
                        const int n0 = wid * 16;
                        v8f za = vzero8();
                        for (int k0 = 0; k0 < H_SZ; k0 += 32) {
                            v16bf a = load_A_bf16(s2, H_SZ, k0, lane);
                            za = wmma_bf16(a, load_B_w(W1, H_SZ, n0, k0, lane), za);
                        }
#pragma unroll
                        for (int r = 0; r < 8; ++r) {
                            const int m = r + 8 * half;
                            const int col = n0 + nlo;
                            zbuf[m * OH_SZ + col] = f2bf(tanhf(za[r] + b1[col]));
                        }
                    }
                    __syncthreads();
                    // k_st = z @ W2^T + b2: 16x512, all 8 waves; A loads hoisted
                    {
                        v16bf a0 = load_A_bf16(zbuf, OH_SZ, 0, lane);
                        v16bf a1 = load_A_bf16(zbuf, OH_SZ, 32, lane);
#pragma unroll
                        for (int nb = 0; nb < 4; ++nb) {
                            const int n0 = (wid * 4 + nb) * 16;
                            v8f ka = vzero8();
                            ka = wmma_bf16(a0, load_B_w(W2, OH_SZ, n0, 0, lane), ka);
                            ka = wmma_bf16(a1, load_B_w(W2, OH_SZ, n0, 32, lane), ka);
#pragma unroll
                            for (int r = 0; r < 8; ++r) {
                                const int m = r + 8 * half;
                                const int col = n0 + nlo;
                                kbuf[st * NE + m * H_SZ + col] = f2bf(ka[r] + b2[col]);
                            }
                        }
                    }
                    __syncthreads();
                }
                // h += dt * sum_j b_j k_j   (b[1] == 0)
                for (int i = tid; i < NE; i += NTHREADS) {
                    float dt = dts[i >> 9];
                    float a2 = c_DPB[0] * bf2f(kbuf[0 * NE + i]) +
                               c_DPB[2] * bf2f(kbuf[2 * NE + i]) +
                               c_DPB[3] * bf2f(kbuf[3 * NE + i]) +
                               c_DPB[4] * bf2f(kbuf[4 * NE + i]) +
                               c_DPB[5] * bf2f(kbuf[5 * NE + i]);
                    s1[i] += dt * a2;
                }
                __syncthreads();
            }

            // commit h_l (bf16 mirror); layer-1 h is the timestep output;
            // at t == T-1 also emit stacked hT directly from fp32 s1
            for (int i = tid; i < NE; i += NTHREADS) {
                float v = s1[i];
                hbfl[i] = f2bf(v);
                int row = i >> 9, col = i & (H_SZ - 1);
                if (l == 1)
                    out[((size_t)(rbase + row) * T_SZ + t) * H_SZ + col] = v;
                if (t == T_SZ - 1)
                    outHT[(size_t)l * B_SZ * H_SZ + (size_t)(rbase + row) * H_SZ + col] = v;
            }
            __syncthreads();
        }
    }

    // final stacked cT (2,B,H); hT already written at t == T-1
    for (int i = tid; i < NE; i += NTHREADS) {
        int row = i >> 9, col = i & (H_SZ - 1);
        size_t o = (size_t)(rbase + row) * H_SZ + col;
        outCT[o] = c0[i];
        outCT[(size_t)B_SZ * H_SZ + o] = c1[i];
    }
}

extern "C" void kernel_launch(void* const* d_in, const int* in_sizes, int n_in,
                              void* d_out, int out_size, void* d_ws, size_t ws_size,
                              hipStream_t stream) {
    const float* x = (const float*)d_in[0];
    const float* tau = (const float*)d_in[1];
    const float* fWi0 = (const float*)d_in[2];
    const float* fWh0 = (const float*)d_in[3];
    const float* bi0 = (const float*)d_in[4];
    const float* bh0 = (const float*)d_in[5];
    const float* fW10 = (const float*)d_in[6];
    const float* b1_0 = (const float*)d_in[7];
    const float* fW20 = (const float*)d_in[8];
    const float* b2_0 = (const float*)d_in[9];
    const float* fWi1 = (const float*)d_in[10];
    const float* fWh1 = (const float*)d_in[11];
    const float* bi1 = (const float*)d_in[12];
    const float* bh1 = (const float*)d_in[13];
    const float* fW11 = (const float*)d_in[14];
    const float* b1_1 = (const float*)d_in[15];
    const float* fW21 = (const float*)d_in[16];
    const float* b2_1 = (const float*)d_in[17];

    // workspace layout: bf16 weights then fp32 bias sums
    __bf16* wsb = (__bf16*)d_ws;
    const int nWi0 = 4 * H_SZ * IN_SZ;  // 524288
    const int nWh = 4 * H_SZ * H_SZ;    // 1048576
    const int nW1 = OH_SZ * H_SZ;       // 32768
    __bf16* Wi0 = wsb;
    __bf16* Wh0 = Wi0 + nWi0;
    __bf16* Wi1 = Wh0 + nWh;
    __bf16* Wh1 = Wi1 + nWh;
    __bf16* W1_0 = Wh1 + nWh;
    __bf16* W2_0 = W1_0 + nW1;
    __bf16* W1_1 = W2_0 + nW1;
    __bf16* W2_1 = W1_1 + nW1;
    float* bsum0 = (float*)(W2_1 + nW1);
    float* bsum1 = bsum0 + 4 * H_SZ;

    auto cvt = [&](const float* src, __bf16* dst, int n) {
        cvt_bf16_kernel<<<(n + 255) / 256, 256, 0, stream>>>(src, dst, n);
    };
    cvt(fWi0, Wi0, nWi0);
    cvt(fWh0, Wh0, nWh);
    cvt(fWi1, Wi1, nWh);
    cvt(fWh1, Wh1, nWh);
    cvt(fW10, W1_0, nW1);
    cvt(fW20, W2_0, nW1);
    cvt(fW11, W1_1, nW1);
    cvt(fW21, W2_1, nW1);
    bias_sum_kernel<<<8, 256, 0, stream>>>(bi0, bh0, bsum0, 4 * H_SZ);
    bias_sum_kernel<<<8, 256, 0, stream>>>(bi1, bh1, bsum1, 4 * H_SZ);

    float* out = (float*)d_out;
    float* outHT = out + (size_t)B_SZ * T_SZ * H_SZ;
    float* outCT = outHT + (size_t)2 * B_SZ * H_SZ;

    const int NE = WG_ROWS * H_SZ;
    size_t lds = (size_t)3 * NE * 4              // c0, c1, s1 fp32
                 + (size_t)(2 * NE) * 2          // hbf0, hbf1
                 + (size_t)(NE + 6 * NE) * 2     // s2 + 6 k-slots
                 + (size_t)WG_ROWS * OH_SZ * 2   // zbuf
                 + WG_ROWS * 4;                  // dts
    odelstm_fused<<<B_SZ / WG_ROWS, NTHREADS, lds, stream>>>(
        x, tau, Wi0, Wh0, Wi1, Wh1, W1_0, W2_0, W1_1, W2_1, bsum0, bsum1, b1_0, b2_0,
        b1_1, b2_1, out, outHT, outCT);
}
